// MoE_40956808135216
// MI455X (gfx1250) — compile-verified, matrix operand order checked by reading, not run on hardware
//
#include <hip/hip_runtime.h>
#include <hip/hip_bf16.h>
#include <math.h>

// ---------------- problem constants (match reference) ----------------
#define E_EXP   8
#define TOP_K   2
#define ALPHA   1.702f
#define LIMIT   7.0f
#define D_DIM   1024
#define F_DIM   1024
#define T_TOK   2048            // B*S
#define TT32    (T_TOK / 32)    // 32-token tiles

// ---------------- WMMA types ----------------
typedef __bf16 v16bf __attribute__((ext_vector_type(16)));
typedef __bf16 v8bf  __attribute__((ext_vector_type(8)));
typedef float  v8f   __attribute__((ext_vector_type(8)));
typedef float  v4f   __attribute__((ext_vector_type(4)));

__device__ __forceinline__ v8f wmma_bf16(v16bf a, v16bf b, v8f c) {
    return __builtin_amdgcn_wmma_f32_16x16x32_bf16(
        /*neg_a=*/false, a, /*neg_b=*/false, b,
        /*c_mod=*/(short)0, c, /*reuse_a=*/false, /*reuse_b=*/false);
}

union BF16x16 { v16bf v; v8bf h[2]; };

// 16-bit A/B frag layout (ISA 7.12.2): lane half kh=0 -> K {k0..k0+7, k0+16..k0+23},
// kh=1 -> K {k0+8..k0+15, k0+24..k0+31}. Two contiguous 16B chunks per lane.
__device__ __forceinline__ v16bf load_frag_bf16(const __bf16* __restrict__ row,
                                                int k0, int kh) {
    BF16x16 u;
    u.h[0] = *(const v8bf*)(row + k0 + kh * 8);
    u.h[1] = *(const v8bf*)(row + k0 + 16 + kh * 8);
    return u.v;
}

__device__ __forceinline__ v8bf cvt8(v4f a, v4f b) {
    v8bf r;
    r[0] = (__bf16)a[0]; r[1] = (__bf16)a[1]; r[2] = (__bf16)a[2]; r[3] = (__bf16)a[3];
    r[4] = (__bf16)b[0]; r[5] = (__bf16)b[1]; r[6] = (__bf16)b[2]; r[7] = (__bf16)b[3];
    return r;
}

// fp32 weights stay L2-resident (192MB covers the whole model); cvt in-register.
__device__ __forceinline__ v16bf load_frag_f32(const float* __restrict__ row,
                                               int k0, int kh) {
    const float* q0 = row + k0 + kh * 8;
    const float* q1 = q0 + 16;
    BF16x16 u;
    u.h[0] = cvt8(*(const v4f*)q0, *(const v4f*)(q0 + 4));
    u.h[1] = cvt8(*(const v4f*)q1, *(const v4f*)(q1 + 4));
    return u.v;
}

// ---------------- kernel 0: zero expert counters ----------------
__global__ void zero_cnt_kernel(int* __restrict__ cnt) {
    if (threadIdx.x < E_EXP) cnt[threadIdx.x] = 0;
}

// ---------------- kernel 1: x fp32 -> bf16 ----------------
__global__ void cvt_x_kernel(const float* __restrict__ x, __bf16* __restrict__ xb) {
    size_t i = ((size_t)blockIdx.x * blockDim.x + threadIdx.x) * 4;
    v4f v = *(const v4f*)(x + i);
    xb[i + 0] = (__bf16)v[0]; xb[i + 1] = (__bf16)v[1];
    xb[i + 2] = (__bf16)v[2]; xb[i + 3] = (__bf16)v[3];
}

// ---------------- kernel 2: gate + top-2 routing (1 wave / token) ----------------
__global__ void gate_kernel(const float* __restrict__ x,
                            const float* __restrict__ gw,
                            const float* __restrict__ gb,
                            int* __restrict__ cnt,
                            int* __restrict__ toks,
                            float* __restrict__ wts) {
    const int t    = blockIdx.x * 8 + (threadIdx.x >> 5);
    const int lane = threadIdx.x & 31;
    float acc[E_EXP];
#pragma unroll
    for (int e = 0; e < E_EXP; ++e) acc[e] = 0.0f;
    const float* xp = x + (size_t)t * D_DIM + lane;
    for (int j = 0; j < D_DIM / 32; ++j) {
        float xv = xp[j * 32];
#pragma unroll
        for (int e = 0; e < E_EXP; ++e)
            acc[e] += xv * gw[e * D_DIM + j * 32 + lane];
    }
#pragma unroll
    for (int e = 0; e < E_EXP; ++e)
        for (int off = 16; off > 0; off >>= 1)
            acc[e] += __shfl_xor(acc[e], off, 32);
    if (lane == 0) {
        float mx = acc[0];
#pragma unroll
        for (int e = 1; e < E_EXP; ++e) mx = fmaxf(mx, acc[e]);
        float ex[E_EXP], s = 0.0f;
#pragma unroll
        for (int e = 0; e < E_EXP; ++e) { ex[e] = __expf(acc[e] - mx); s += ex[e]; }
        float inv = 1.0f / s;
        float score[E_EXP], biased[E_EXP];
#pragma unroll
        for (int e = 0; e < E_EXP; ++e) {
            score[e]  = ex[e] * inv;
            biased[e] = score[e] + gb[e];
        }
        int top1 = 0;
#pragma unroll
        for (int e = 1; e < E_EXP; ++e) if (biased[e] > biased[top1]) top1 = e;
        int top2 = (top1 == 0) ? 1 : 0;
#pragma unroll
        for (int e = 0; e < E_EXP; ++e)
            if (e != top1 && biased[e] > biased[top2]) top2 = e;
        int sel[TOP_K] = { top1, top2 };
#pragma unroll
        for (int k = 0; k < TOP_K; ++k) {
            int e = sel[k];
            int slot = atomicAdd(&cnt[e], 1);
            toks[e * T_TOK + slot] = t;
            wts[e * T_TOK + slot]  = score[e];   // ROUTE_SCALE == 1.0
        }
    }
}

// ---------------- kernel 3: up-proj (w1,w3) + SwiGLU -> act (bf16) ----------------
// block = 128 threads = 4 waves; wave computes a 32-token x 16-F tile:
// two A frags share every B frag (halves fp32 weight loads + cvt per WMMA).
// Interleaved split: h cols 2f (glu) / 2f+1 (linear) for w1 and w3 -> 8 accums.
template <bool SHARED>
__global__ void up_act_kernel(const __bf16* __restrict__ xb,
                              const float* __restrict__ w1,
                              const float* __restrict__ b1,
                              const float* __restrict__ w3,
                              const float* __restrict__ b3,
                              const int* __restrict__ cnt,
                              const int* __restrict__ toks,
                              __bf16* __restrict__ act) {
    const int wave = threadIdx.x >> 5, lane = threadIdx.x & 31;
    const int kh = lane >> 4, lm = lane & 15;
    int e, tt;
    if (SHARED) { e = 0; tt = blockIdx.x; }
    else        { e = blockIdx.x / TT32; tt = blockIdx.x % TT32; }
    const int n_tok = SHARED ? T_TOK : cnt[e];
    if (tt * 32 >= n_tok) return;                       // block-uniform exit

    // A rows: m-tile 0 -> tokens tt*32+lm, m-tile 1 -> +16 (clamp padded slots)
    int s0 = tt * 32 + lm,  s1 = s0 + 16;
    int r0 = (s0 < n_tok) ? s0 : (n_tok - 1);
    int r1 = (s1 < n_tok) ? s1 : (n_tok - 1);
    int tok0 = SHARED ? r0 : toks[e * T_TOK + r0];
    int tok1 = SHARED ? r1 : toks[e * T_TOK + r1];
    const __bf16* ap0 = xb + (size_t)tok0 * D_DIM;
    const __bf16* ap1 = xb + (size_t)tok1 * D_DIM;

    const int f0 = (blockIdx.y * 4 + wave) * 16;        // F tile base
    const size_t wbase = SHARED ? 0 : (size_t)e * (2 * F_DIM) * D_DIM;
    const float* p1e = w1 + wbase + (size_t)(2 * (f0 + lm)) * D_DIM;  // even col
    const float* p1o = p1e + D_DIM;                                   // odd col
    const float* p3e = w3 + wbase + (size_t)(2 * (f0 + lm)) * D_DIM;
    const float* p3o = p3e + D_DIM;

    v8f c1e0 = {}, c1e1 = {}, c1o0 = {}, c1o1 = {};
    v8f c3e0 = {}, c3e1 = {}, c3o0 = {}, c3o1 = {};
    for (int ks = 0; ks < D_DIM / 32; ++ks) {
        const int k0 = ks * 32;
        __builtin_prefetch(p1e + k0 + 64, 0, 1);
        v16bf a0 = load_frag_bf16(ap0, k0, kh);
        v16bf a1 = load_frag_bf16(ap1, k0, kh);
        v16bf b;
        b = load_frag_f32(p1e, k0, kh);
        c1e0 = wmma_bf16(a0, b, c1e0);  c1e1 = wmma_bf16(a1, b, c1e1);
        b = load_frag_f32(p1o, k0, kh);
        c1o0 = wmma_bf16(a0, b, c1o0);  c1o1 = wmma_bf16(a1, b, c1o1);
        b = load_frag_f32(p3e, k0, kh);
        c3e0 = wmma_bf16(a0, b, c3e0);  c3e1 = wmma_bf16(a1, b, c3e1);
        b = load_frag_f32(p3o, k0, kh);
        c3o0 = wmma_bf16(a0, b, c3o0);  c3o1 = wmma_bf16(a1, b, c3o1);
    }

    const size_t bb = SHARED ? 0 : (size_t)e * (2 * F_DIM);
    const float bi1e = b1[bb + 2 * (f0 + lm)],  bi1o = b1[bb + 2 * (f0 + lm) + 1];
    const float bi3e = b3[bb + 2 * (f0 + lm)],  bi3o = b3[bb + 2 * (f0 + lm) + 1];
    const size_t arow_base = SHARED ? 0 : (size_t)e * T_TOK;
#pragma unroll
    for (int mt = 0; mt < 2; ++mt) {
        const v8f& v1e = mt ? c1e1 : c1e0;
        const v8f& v1o = mt ? c1o1 : c1o0;
        const v8f& v3e = mt ? c3e1 : c3e0;
        const v8f& v3o = mt ? c3o1 : c3o0;
#pragma unroll
        for (int r = 0; r < 8; ++r) {
            const int sm = tt * 32 + mt * 16 + r + kh * 8;
            if (sm >= n_tok) continue;
            float hE = (v1e[r] + bi1e) * (v3e[r] + bi3e);
            float hO = (v1o[r] + bi1o) * (v3o[r] + bi3o);
            float g  = fminf(hE, LIMIT);
            float lv = fminf(fmaxf(hO, -LIMIT), LIMIT);
            float sg = 1.0f / (1.0f + __expf(-ALPHA * g));
            float av = g * sg * (lv + 1.0f);
            act[(arow_base + sm) * F_DIM + f0 + lm] = (__bf16)av;
        }
    }
}

// ---------------- kernel 4: down-proj (w2), 32-token x 16-d tiles ----------------
// SHARED: out[t][d] = act_s @ sw2^T + sb2   (initializes output)
// routed: out[tok][d] += wt * (act @ w2[e]^T + b2[e])   via global_atomic_add_f32
template <bool SHARED>
__global__ void down_proj_kernel(const __bf16* __restrict__ act,
                                 const float* __restrict__ w2,
                                 const float* __restrict__ b2,
                                 const int* __restrict__ cnt,
                                 const int* __restrict__ toks,
                                 const float* __restrict__ wts,
                                 float* __restrict__ out) {
    const int wave = threadIdx.x >> 5, lane = threadIdx.x & 31;
    const int kh = lane >> 4, lm = lane & 15;
    int e, tt;
    if (SHARED) { e = 0; tt = blockIdx.x; }
    else        { e = blockIdx.x / TT32; tt = blockIdx.x % TT32; }
    const int n_tok = SHARED ? T_TOK : cnt[e];
    if (tt * 32 >= n_tok) return;

    int s0 = tt * 32 + lm,  s1 = s0 + 16;
    int r0 = (s0 < n_tok) ? s0 : (n_tok - 1);
    int r1 = (s1 < n_tok) ? s1 : (n_tok - 1);
    const size_t arow_base = SHARED ? 0 : (size_t)e * T_TOK;
    const __bf16* ap0 = act + (arow_base + r0) * F_DIM;
    const __bf16* ap1 = act + (arow_base + r1) * F_DIM;

    const int d0 = (blockIdx.y * 4 + wave) * 16;
    const float* bp = w2 + (SHARED ? 0 : (size_t)e * D_DIM * F_DIM)
                         + (size_t)(d0 + lm) * F_DIM;

    v8f c0 = {}, c1 = {};
    for (int ks = 0; ks < F_DIM / 32; ++ks) {
        const int k0 = ks * 32;
        __builtin_prefetch(bp + k0 + 64, 0, 1);
        v16bf b  = load_frag_f32(bp, k0, kh);
        c0 = wmma_bf16(load_frag_bf16(ap0, k0, kh), b, c0);
        c1 = wmma_bf16(load_frag_bf16(ap1, k0, kh), b, c1);
    }

    const float bias = b2[(SHARED ? 0 : (size_t)e * D_DIM) + d0 + lm];
#pragma unroll
    for (int mt = 0; mt < 2; ++mt) {
        const v8f& c = mt ? c1 : c0;
#pragma unroll
        for (int r = 0; r < 8; ++r) {
            const int sm = tt * 32 + mt * 16 + r + kh * 8;
            if (sm >= n_tok) continue;
            float v = c[r] + bias;
            if (SHARED) {
                out[(size_t)sm * D_DIM + d0 + lm] = v;
            } else {
                int   tokm = toks[e * T_TOK + sm];
                float wt   = wts[e * T_TOK + sm];
                atomicAdd(&out[(size_t)tokm * D_DIM + d0 + lm], wt * v);
            }
        }
    }
}

// ---------------- launcher ----------------
extern "C" void kernel_launch(void* const* d_in, const int* in_sizes, int n_in,
                              void* d_out, int out_size, void* d_ws, size_t ws_size,
                              hipStream_t stream) {
    (void)in_sizes; (void)n_in; (void)out_size; (void)ws_size;
    const float* x      = (const float*)d_in[0];
    const float* gate_w = (const float*)d_in[1];
    const float* gate_b = (const float*)d_in[2];
    const float* w1     = (const float*)d_in[3];
    const float* b1     = (const float*)d_in[4];
    const float* w3     = (const float*)d_in[5];
    const float* b3     = (const float*)d_in[6];
    const float* w2     = (const float*)d_in[7];
    const float* b2     = (const float*)d_in[8];
    const float* sw1    = (const float*)d_in[9];
    const float* sb1    = (const float*)d_in[10];
    const float* sw3    = (const float*)d_in[11];
    const float* sb3    = (const float*)d_in[12];
    const float* sw2    = (const float*)d_in[13];
    const float* sb2    = (const float*)d_in[14];
    float* out = (float*)d_out;

    // workspace carve-up
    char* ws = (char*)d_ws;
    int*    cnt   = (int*)(ws);                                  // 256 B
    int*    toks  = (int*)(ws + 256);                            // E*T*4
    float*  wts   = (float*)(ws + 256 + E_EXP * T_TOK * 4);      // E*T*4
    size_t  off   = 256 + (size_t)E_EXP * T_TOK * 8;
    __bf16* xb    = (__bf16*)(ws + off);            off += (size_t)T_TOK * D_DIM * 2;
    __bf16* act   = (__bf16*)(ws + off);            off += (size_t)E_EXP * T_TOK * F_DIM * 2;
    __bf16* act_s = (__bf16*)(ws + off);

    zero_cnt_kernel<<<1, 32, 0, stream>>>(cnt);
    cvt_x_kernel<<<(T_TOK * D_DIM) / (256 * 4), 256, 0, stream>>>(x, xb);
    gate_kernel<<<T_TOK / 8, 256, 0, stream>>>(x, gate_w, gate_b, cnt, toks, wts);

    // shared expert (also initializes out)
    up_act_kernel<true><<<dim3(TT32, F_DIM / 64), 128, 0, stream>>>(
        xb, sw1, sb1, sw3, sb3, nullptr, nullptr, act_s);
    down_proj_kernel<true><<<dim3(TT32, D_DIM / 64), 128, 0, stream>>>(
        act_s, sw2, sb2, nullptr, nullptr, nullptr, out);

    // routed experts (gathered token tiles; inactive tiles exit early)
    up_act_kernel<false><<<dim3(E_EXP * TT32, F_DIM / 64), 128, 0, stream>>>(
        xb, w1, b1, w3, b3, cnt, toks, act);
    down_proj_kernel<false><<<dim3(E_EXP * TT32, D_DIM / 64), 128, 0, stream>>>(
        act, w2, b2, cnt, toks, wts, out);
}
